// EnhanceModule_27754078667539
// MI455X (gfx1250) — matrix operand order repaired and checked by reading, not compiled, not run on hardware
//
#include <hip/hip_runtime.h>
#include <cstdint>

// ---------------------------------------------------------------------------
// MI455X (gfx1250) implementation of the EnhanceModule reference.
//  * channel-mixing 3x3 convs  -> implicit-GEMM on v_wmma_f32_16x16x32_f16
//  * depthwise gauss / resize / bilateral / einsums -> fused VALU kernels
//  * wave32: one 16x16 WMMA tile per wave, 8 waves per 256-thread block
// ---------------------------------------------------------------------------

typedef _Float16 f16;
typedef _Float16 v16h __attribute__((ext_vector_type(16)));
typedef _Float16 v8h  __attribute__((ext_vector_type(8)));
typedef float    v8f  __attribute__((ext_vector_type(8)));

#define DEV __device__ __forceinline__

DEV float warp_sum(float v) {
#pragma unroll
  for (int o = 16; o > 0; o >>= 1) v += __shfl_down(v, o, 32);
  return v;
}
DEV unsigned warp_max_u(unsigned v) {
#pragma unroll
  for (int o = 16; o > 0; o >>= 1) {
    unsigned t = __shfl_down(v, o, 32);
    v = t > v ? t : v;
  }
  return v;
}

// ---------------------------------------------------------------------------
// f32 -> f16 weight pack
// ---------------------------------------------------------------------------
__global__ void k_cvt_f16(const float* __restrict__ s, f16* __restrict__ d, int n) {
  int i = blockIdx.x * 256 + threadIdx.x;
  if (i < n) d[i] = (f16)s[i];
}

// ---------------------------------------------------------------------------
// Implicit-GEMM 3x3 conv (pad=1), stride 1/2, optional scale / residual / act.
//   x: (B,Cin,Hin,Win) f32    wgt: (Cout, Cin*9) f16    y: (B,Cout,Hout,Wout)
// Block: 256 threads (8 waves). Tile: 64 output pixels x 32 output channels.
// K loop over Cin*9 in chunks of 64 -> two back-to-back WMMAs per stage.
// LDS tiles padded to a 16B-multiple pitch so every lane's fragment is two
// contiguous ds_load_b128s (A stored [m][k], B stored transposed [n][k]).
// act: 0=none 1=relu 2=sigmoid
// ---------------------------------------------------------------------------
__global__ void __launch_bounds__(256)
k_conv3x3_wmma(const float* __restrict__ x, const f16* __restrict__ wgt,
               const float* __restrict__ bias, float* __restrict__ y,
               const float* __restrict__ res,
               int B, int Cin, int Hin, int Win, int Hout, int Wout, int Cout,
               int stride, int act, float scale) {
  const int NP = B * Hout * Wout;
  const int K  = Cin * 9;
  const int pixBase  = blockIdx.x * 64;
  const int coutBase = blockIdx.y * 32;
  const int tid  = threadIdx.x;
  const int lane = tid & 31;
  const int wid  = tid >> 5;
  const int wm = wid & 3;   // pixel sub-tile (4 x 16 pixels)
  const int wn = wid >> 2;  // channel sub-tile (2 x 16 channels)
  const int ml = lane & 15;
  const int hh = lane >> 4;

  __shared__ __align__(16) f16 As[64][72];   // [pixel][k]  (pitch 144B)
  __shared__ __align__(16) f16 Bs[32][72];   // [cout][k]   (transposed stage)
  __shared__ int pB[64], pY[64], pX[64];     // per-pixel decode (once/block)

  if (tid < 64) {
    int p = pixBase + tid;
    int b = 0, oy = 0, ox = 0;
    bool ok = p < NP;
    if (ok) {
      b = p / (Hout * Wout);
      int rp = p - b * Hout * Wout;
      oy = rp / Wout;
      ox = rp - oy * Wout;
    }
    pB[tid] = ok ? b : -1;
    pY[tid] = oy * stride - 1;
    pX[tid] = ox * stride - 1;
  }
  __syncthreads();

  v8f acc = {0.f, 0.f, 0.f, 0.f, 0.f, 0.f, 0.f, 0.f};

  for (int kc = 0; kc < K; kc += 64) {
    // L2-resident weight stream hint for the next chunk (global_prefetch_b8)
    __builtin_prefetch(&wgt[(long)coutBase * K + kc + 64], 0, 1);

    // ---- stage A chunk: 64 pixels x 64 k values (im2col gather, zero pad)
#pragma unroll
    for (int e = 0; e < 16; ++e) {
      int l  = tid * 16 + e;          // 0..4095
      int m  = l >> 6;
      int kk = l & 63;
      int k  = kc + kk;
      float v = 0.f;
      int b = pB[m];
      if (k < K && b >= 0) {
        int ci = k / 9, r9 = k - ci * 9;
        int kh = r9 / 3, kw2 = r9 - kh * 3;
        int iy = pY[m] + kh;
        int ix = pX[m] + kw2;
        if (iy >= 0 && iy < Hin && ix >= 0 && ix < Win)
          v = x[(((long)b * Cin + ci) * Hin + iy) * Win + ix];
      }
      As[m][kk] = (f16)v;
    }
    // ---- stage B chunk (transposed): 32 cout x 64 k
#pragma unroll
    for (int e = 0; e < 8; ++e) {
      int l  = tid * 8 + e;           // 0..2047
      int n  = l >> 6;
      int kk = l & 63;
      int k  = kc + kk;
      int co = coutBase + n;
      f16 v = (f16)0.f;
      if (k < K && co < Cout) v = wgt[(long)co * K + k];
      Bs[n][kk] = v;
    }
    __syncthreads();

    // ---- two WMMAs per stage; fragments = 2x ds_load_b128 each
    const v8h* arow = (const v8h*)&As[(wm << 4) + ml][0];  // 9 v8h per row
    const v8h* brow = (const v8h*)&Bs[(wn << 4) + ml][0];
#pragma unroll
    for (int q = 0; q < 2; ++q) {
      // A elem i -> k = 32q + (i/8)*16 + hh*8 + i%8  => v8h idx 4q+hh, 4q+2+hh
      v8h a0 = arow[4 * q + hh];
      v8h a1 = arow[4 * q + 2 + hh];
      // B elem i -> k = 32q + (i/8)*16 + hh*8 + i%8  => v8h idx 4q+hh, 4q+2+hh
      v8h b0 = brow[4 * q + hh];
      v8h b1 = brow[4 * q + 2 + hh];
      v16h av = __builtin_shufflevector(a0, a1, 0, 1, 2, 3, 4, 5, 6, 7,
                                        8, 9, 10, 11, 12, 13, 14, 15);
      v16h bv = __builtin_shufflevector(b0, b1, 0, 1, 2, 3, 4, 5, 6, 7,
                                        8, 9, 10, 11, 12, 13, 14, 15);
      acc = __builtin_amdgcn_wmma_f32_16x16x32_f16(false, av, false, bv,
                                                   (short)0, acc, false, false);
    }
    __syncthreads();
  }

  // ---- epilogue: D layout lane%16 = N, vgpr r -> M = r + 8*(lane/16)
  int n = coutBase + (wn << 4) + ml;
#pragma unroll
  for (int r = 0; r < 8; ++r) {
    int m = (wm << 4) + r + (hh << 3);
    int p = pixBase + m;
    if (p < NP && n < Cout) {
      int b  = p / (Hout * Wout);
      int rp = p - b * Hout * Wout;
      int oy = rp / Wout, ox = rp - oy * Wout;
      float v = acc[r] + bias[n];
      v *= scale;
      long oidx = (((long)b * Cout + n) * Hout + oy) * Wout + ox;
      if (res) v += res[oidx];
      if (act == 1) v = v > 0.f ? v : 0.f;
      else if (act == 2) v = 1.f / (1.f + __expf(-v));
      y[oidx] = v;
    }
  }
}

// ---------------------------------------------------------------------------
// Depthwise 5x5 gaussian blur (zero pad), optional decimation and fused adds:
//   dst = sgn * blur(src)[oy*decim, ox*decim] + add1 + add2
// Gaussian weights are compile-time constants (sigma=1, normalized).
// ---------------------------------------------------------------------------
__global__ void k_blur5(const float* __restrict__ src, float* __restrict__ dst,
                        const float* __restrict__ add1, const float* __restrict__ add2,
                        float sgn, int B, int C, int Hs, int Ws,
                        int Ho, int Wo, int decim) {
  long i = (long)blockIdx.x * 256 + threadIdx.x;
  long tot = (long)B * C * Ho * Wo;
  if (i >= tot) return;
  int ox = (int)(i % Wo);
  long t = i / Wo;
  int oy = (int)(t % Ho);
  t /= Ho;                              // t = b*C + c
  const float* sp = src + t * (long)Hs * Ws;
  int cy = oy * decim, cx = ox * decim;

  const float wy[5] = {0.13533528323661270f, 0.60653065971263342f, 1.f,
                       0.60653065971263342f, 0.13533528323661270f};
  const float s1 = 0.13533528323661270f * 2.f + 0.60653065971263342f * 2.f + 1.f;
  const float norm = 1.f / (s1 * s1);

  float num = 0.f;
#pragma unroll
  for (int u = 0; u < 5; ++u) {
    int yy = cy + u - 2;
    if (yy < 0 || yy >= Hs) continue;
    float rowa = 0.f;
#pragma unroll
    for (int v = 0; v < 5; ++v) {
      int xx = cx + v - 2;
      if (xx < 0 || xx >= Ws) continue;
      rowa += wy[v] * sp[(long)yy * Ws + xx];
    }
    num += wy[u] * rowa;
  }
  float out = sgn * num * norm;
  if (add1) out += add1[i];
  if (add2) out += add2[i];
  dst[i] = out;
}

// ---------------------------------------------------------------------------
// Bilinear resize (jax.image.resize, antialias=False, half-pixel centers).
// dst may be a channel slice of a larger tensor: dst already offset to its
// channel start; per-batch stride passed explicitly.
// ---------------------------------------------------------------------------
__global__ void k_resize(const float* __restrict__ src, float* __restrict__ dst,
                         int B, int C, int Hi, int Wi, int Ho, int Wo,
                         long dstBatchStride) {
  long i = (long)blockIdx.x * 256 + threadIdx.x;
  long tot = (long)B * C * Ho * Wo;
  if (i >= tot) return;
  int ox = (int)(i % Wo);
  long t = i / Wo;
  int oy = (int)(t % Ho); t /= Ho;
  int c = (int)(t % C);
  int b = (int)(t / C);

  float sy = (float)Hi / (float)Ho, sx = (float)Wi / (float)Wo;
  float fy = (oy + 0.5f) * sy - 0.5f;
  float fx = (ox + 0.5f) * sx - 0.5f;
  int y0 = (int)floorf(fy), x0 = (int)floorf(fx);
  float ay = fy - y0, ax = fx - x0;
  int y1 = y0 + 1, x1 = x0 + 1;
  y0 = y0 < 0 ? 0 : (y0 >= Hi ? Hi - 1 : y0);
  y1 = y1 < 0 ? 0 : (y1 >= Hi ? Hi - 1 : y1);
  x0 = x0 < 0 ? 0 : (x0 >= Wi ? Wi - 1 : x0);
  x1 = x1 < 0 ? 0 : (x1 >= Wi ? Wi - 1 : x1);

  const float* sp = src + ((long)b * C + c) * Hi * Wi;
  float v00 = sp[(long)y0 * Wi + x0], v01 = sp[(long)y0 * Wi + x1];
  float v10 = sp[(long)y1 * Wi + x0], v11 = sp[(long)y1 * Wi + x1];
  float v = (1.f - ay) * ((1.f - ax) * v00 + ax * v01) +
            ay * ((1.f - ax) * v10 + ax * v11);
  dst[(long)b * dstBatchStride + (long)c * Ho * Wo + (long)oy * Wo + ox] = v;
}

// ---------------------------------------------------------------------------
// Bilateral 5x5, edge padding. sig_s = pr[b*8+2+c], sig_c = pr[b*8+5+c].
// ---------------------------------------------------------------------------
__global__ void k_bilateral(const float* __restrict__ x, const float* __restrict__ pr,
                            float* __restrict__ out, int B, int C, int H, int W) {
  long i = (long)blockIdx.x * 256 + threadIdx.x;
  long tot = (long)B * C * H * W;
  if (i >= tot) return;
  int px = (int)(i % W);
  long t = i / W;
  int py = (int)(t % H); t /= H;
  int c = (int)(t % C);
  int b = (int)(t / C);

  const float* sp = x + ((long)b * C + c) * H * W;
  float ss = pr[b * 8 + 2 + c];
  float sc = pr[b * 8 + 5 + c];
  float inv2ss = 1.f / (2.f * ss * ss);
  float inv2sc = 1.f / (2.f * sc * sc + 1e-6f);
  float center = sp[(long)py * W + px];

  float num = 0.f, den = 0.f;
#pragma unroll
  for (int u = 0; u < 5; ++u) {
    int yy = py + u - 2;
    yy = yy < 0 ? 0 : (yy >= H ? H - 1 : yy);
#pragma unroll
    for (int v = 0; v < 5; ++v) {
      int xx = px + v - 2;
      xx = xx < 0 ? 0 : (xx >= W ? W - 1 : xx);
      float p = sp[(long)yy * W + xx];
      float d2 = (float)((u - 2) * (u - 2) + (v - 2) * (v - 2));
      float d = p - center;
      float tw = __expf(-d2 * inv2ss) * __expf(-d * d * inv2sc);
      num += p * tw;
      den += tw;
    }
  }
  den = den > 1e-6f ? den : 1e-6f;
  out[i] = num / den;
}

// ---------------------------------------------------------------------------
// max(img) flag + conditional 1/255 prescale (jnp.where(max(x)>1, x/255, x))
// ---------------------------------------------------------------------------
__global__ void k_zero_u32(unsigned* p) {
  if (blockIdx.x == 0 && threadIdx.x == 0) *p = 0u;
}
__global__ void k_maxflag(const float* __restrict__ x, unsigned* flag, int n) {
  int i = blockIdx.x * 256 + threadIdx.x;
  float v = (i < n) ? x[i] : 0.f;     // inputs are non-negative images
  unsigned b = __float_as_uint(v < 0.f ? 0.f : v);
  b = warp_max_u(b);
  if ((threadIdx.x & 31) == 0) atomicMax(flag, b);
}
__global__ void k_prescale(const float* __restrict__ x, float* __restrict__ y,
                           const unsigned* __restrict__ flag, int n) {
  int i = blockIdx.x * 256 + threadIdx.x;
  if (i >= n) return;
  float s = (__uint_as_float(*flag) > 1.0f) ? (1.f / 255.f) : 1.f;
  y[i] = x[i] * s;
}

// exposed = x * pr[b*8+0]
__global__ void k_expmul(const float* __restrict__ x, const float* __restrict__ pr,
                         float* __restrict__ y, int B, int CHW) {
  long i = (long)blockIdx.x * 256 + threadIdx.x;
  long tot = (long)B * CHW;
  if (i >= tot) return;
  int b = (int)(i / CHW);
  y[i] = x[i] * pr[b * 8 + 0];
}

__global__ void k_mul(const float* __restrict__ a, const float* __restrict__ b,
                      float* __restrict__ y, long n) {
  long i = (long)blockIdx.x * 256 + threadIdx.x;
  if (i < n) y[i] = a[i] * b[i];
}

// ---------------------------------------------------------------------------
// opp head: (8,128,8,6) full conv on 8x6 input -> pr (B,8)
// one block per (b,o) output, 256-thread reduction over 128*48 elements
// ---------------------------------------------------------------------------
__global__ void __launch_bounds__(256)
k_opp_head(const float* __restrict__ x5, const float* __restrict__ w6,
           const float* __restrict__ b6, float* __restrict__ pr) {
  int bo = blockIdx.x;
  int b = bo >> 3, o = bo & 7;
  const float* xp = x5 + (long)b * 128 * 48;
  const float* wp = w6 + (long)o * 128 * 48;
  float s = 0.f;
  for (int i = threadIdx.x; i < 128 * 48; i += 256) s += xp[i] * wp[i];
  s = warp_sum(s);
  __shared__ float part[8];
  if ((threadIdx.x & 31) == 0) part[threadIdx.x >> 5] = s;
  __syncthreads();
  if (threadIdx.x == 0) {
    float t = 0.f;
    for (int w = 0; w < 8; ++w) t += part[w];
    pr[b * 8 + o] = t + b6[o];
  }
}

// v[b,c,d] = sum_n u[b,c,n] * f[b,d,n]   (grid = B*72 blocks)
__global__ void __launch_bounds__(256)
k_einsum_v(const float* __restrict__ u, const float* __restrict__ f,
           float* __restrict__ v, int N) {
  int id = blockIdx.x;
  int b = id / 72, r = id % 72, c = r / 24, d = r % 24;
  const float* up = u + ((long)b * 3 + c) * N;
  const float* fp = f + ((long)b * 24 + d) * N;
  float s = 0.f;
  for (int i = threadIdx.x; i < N; i += 256) s += up[i] * fp[i];
  s = warp_sum(s);
  __shared__ float part[8];
  if ((threadIdx.x & 31) == 0) part[threadIdx.x >> 5] = s;
  __syncthreads();
  if (threadIdx.x == 0) {
    float t = 0.f;
    for (int w = 0; w < 8; ++w) t += part[w];
    v[id] = t;
  }
}

// res[b,d,n] = sum_c v[b,c,d] * u[b,c,n]
__global__ void k_apply_v(const float* __restrict__ v, const float* __restrict__ u,
                          float* __restrict__ res, int N) {
  long i = (long)blockIdx.x * 256 + threadIdx.x;
  long tot = 8L * 24 * N;
  if (i >= tot) return;
  int n = (int)(i % N);
  long t = i / N;
  int d = (int)(t % 24);
  int b = (int)(t / 24);
  float s = 0.f;
#pragma unroll
  for (int c = 0; c < 3; ++c)
    s += v[b * 72 + c * 24 + d] * u[((long)b * 3 + c) * N + n];
  res[i] = s;
}

// ---------------------------------------------------------------------------
// host orchestration
// ---------------------------------------------------------------------------
extern "C" void kernel_launch(void* const* d_in, const int* in_sizes, int n_in,
                              void* d_out, int out_size, void* d_ws, size_t ws_size,
                              hipStream_t stream) {
  (void)in_sizes; (void)n_in; (void)out_size; (void)ws_size;
  const float BN_S = 0.99999500003749971f;  // 1/sqrt(1+1e-5)
  const int B = 8, H = 256, W = 192;
  const long P1 = (long)B * 3 * H * W;            // 1,179,648

  // ---- input pointers (pytree flatten order: dict keys sorted) ----
  const float* img   = (const float*)d_in[0];
  const float* csf_b = (const float*)d_in[1];
  const float* csf_w = (const float*)d_in[2];
  const float *lg_b[6], *lg_w[6];
  for (int i = 0; i < 6; ++i) { lg_b[i] = (const float*)d_in[3 + i]; lg_w[i] = (const float*)d_in[9 + i]; }
  const float *m1_b[4], *m1_w[4], *m2_b[4], *m2_w[4];
  for (int i = 0; i < 4; ++i) {
    m1_b[i] = (const float*)d_in[15 + i]; m1_w[i] = (const float*)d_in[19 + i];
    m2_b[i] = (const float*)d_in[23 + i]; m2_w[i] = (const float*)d_in[27 + i];
  }
  const float *op_b[6], *op_w[6];
  for (int i = 0; i < 6; ++i) { op_b[i] = (const float*)d_in[31 + i]; op_w[i] = (const float*)d_in[37 + i]; }

  // ---- workspace bump allocator ----
  char* base = (char*)d_ws;
  size_t off = 0;
  auto allocB = [&](size_t bytes) -> char* {
    char* p = base + off;
    off += bytes; off = (off + 255) & ~(size_t)255;
    return p;
  };
  auto cvt = [&](const float* s, f16* d, int n) {
    k_cvt_f16<<<dim3((n + 255) / 256), 256, 0, stream>>>(s, d, n);
  };

  // f16 weight pool
  const int m1sz[4] = {648, 648, 5184, 648};
  f16 *m1wh[4], *m2wh[4];
  for (int i = 0; i < 4; ++i) { m1wh[i] = (f16*)allocB(m1sz[i] * 2); m2wh[i] = (f16*)allocB(m1sz[i] * 2); }
  f16* csfwh = (f16*)allocB(243 * 2);
  f16* lgwh[6];
  for (int i = 0; i < 6; ++i) lgwh[i] = (f16*)allocB(81 * 2);
  const int opsz[5] = {864, 18432, 73728, 147456, 147456};
  f16* opwh[5];
  for (int i = 0; i < 5; ++i) opwh[i] = (f16*)allocB(opsz[i] * 2);

  // float buffers
  float* A24  = (float*)allocB(9437184L * 4);   // 24ch full-res scratch (x / res)
  float* B24  = (float*)allocB(9437184L * 4);   // 24ch full-res scratch (f)
  float* XN   = (float*)allocB(P1 * 4);
  float* G1   = (float*)allocB(294912L * 4);
  float* G2   = (float*)allocB(73728L * 4);
  float* G3   = (float*)allocB(18432L * 4);
  float* H1   = (float*)allocB(P1 * 4);
  float* H2   = (float*)allocB(294912L * 4);
  float* H3   = (float*)allocB(73728L * 4);
  float* TMPF = (float*)allocB(P1 * 4);
  float* U3   = (float*)allocB(P1 * 4);
  float* Y1   = (float*)allocB(P1 * 4);
  float* Y2   = (float*)allocB(294912L * 4);
  float* Y3   = (float*)allocB(73728L * 4);
  float* HH1  = (float*)allocB(P1 * 4);
  float* HH2  = (float*)allocB(294912L * 4);
  float* HH3  = (float*)allocB(73728L * 4);
  float* LL1  = (float*)allocB(294912L * 4);
  float* LL2  = (float*)allocB(73728L * 4);
  float* LL3  = (float*)allocB(18432L * 4);
  float* PR   = (float*)allocB(64 * 4);
  float* VV   = (float*)allocB(576 * 4);
  unsigned* FLAG = (unsigned*)allocB(64);

  // aliases inside A24 (opp runs before the mld phase; concat after it)
  float* OPP1 = A24;
  float* OPP2 = OPP1 + 3145728;
  float* OPP3 = OPP2 + 1572864;
  float* OPP4 = OPP3 + 786432;
  float* OPP5 = OPP4 + 196608;
  float* C1 = A24;                 // 8 x 9 x 256 x 192
  float* C2 = C1 + 3538944;        // 8 x 9 x 128 x 96
  float* C3 = C2 + 884736;         // 8 x 9 x 64 x 48
  // aliases inside B24 (low_freq full-res temps, all dead before mld phase)
  float* EXPO = B24;
  float* LGA = EXPO + P1;
  float* LGB = LGA + P1;
  float* LGC = LGB + P1;
  float* LGD = LGC + P1;
  float* GAMMA = LGD + P1;
  float* BILO = GAMMA + P1;

  auto g1d = [](long n) { return dim3((unsigned)((n + 255) / 256)); };
  auto conv = [&](const float* x, const f16* w, const float* bias, float* y,
                  const float* res, int Cin, int Hin, int Win, int stride,
                  int Cout, int act, float scale) {
    int Ho = (Hin + 2 - 3) / stride + 1;
    int Wo = (Win + 2 - 3) / stride + 1;
    long NP = (long)B * Ho * Wo;
    dim3 g((unsigned)((NP + 63) / 64), (unsigned)((Cout + 31) / 32));
    k_conv3x3_wmma<<<g, 256, 0, stream>>>(x, w, bias, y, res, B, Cin, Hin, Win,
                                          Ho, Wo, Cout, stride, act, scale);
  };

  // ---- pack weights to f16 ----
  for (int i = 0; i < 4; ++i) { cvt(m1_w[i], m1wh[i], m1sz[i]); cvt(m2_w[i], m2wh[i], m1sz[i]); }
  cvt(csf_w, csfwh, 243);
  for (int i = 0; i < 6; ++i) cvt(lg_w[i], lgwh[i], 81);
  for (int i = 0; i < 5; ++i) cvt(op_w[i], opwh[i], opsz[i]);

  // ---- Laplacian pyramid on raw img ----
  k_blur5<<<g1d((long)B * 3 * 128 * 96), 256, 0, stream>>>(img, G1, nullptr, nullptr, 1.f, B, 3, 256, 192, 128, 96, 2);
  k_blur5<<<g1d((long)B * 3 * 64 * 48), 256, 0, stream>>>(G1, G2, nullptr, nullptr, 1.f, B, 3, 128, 96, 64, 48, 2);
  k_blur5<<<g1d((long)B * 3 * 32 * 24), 256, 0, stream>>>(G2, G3, nullptr, nullptr, 1.f, B, 3, 64, 48, 32, 24, 2);
  // H_i = g_i - blur(resize2x(g_{i+1}))
  k_resize<<<g1d(P1), 256, 0, stream>>>(G1, TMPF, B, 3, 128, 96, 256, 192, 3L * 256 * 192);
  k_blur5<<<g1d(P1), 256, 0, stream>>>(TMPF, H1, img, nullptr, -1.f, B, 3, 256, 192, 256, 192, 1);
  k_resize<<<g1d(294912L), 256, 0, stream>>>(G2, TMPF, B, 3, 64, 48, 128, 96, 3L * 128 * 96);
  k_blur5<<<g1d(294912L), 256, 0, stream>>>(TMPF, H2, G1, nullptr, -1.f, B, 3, 128, 96, 128, 96, 1);
  k_resize<<<g1d(73728L), 256, 0, stream>>>(G3, TMPF, B, 3, 32, 24, 64, 48, 3L * 64 * 48);
  k_blur5<<<g1d(73728L), 256, 0, stream>>>(TMPF, H3, G2, nullptr, -1.f, B, 3, 64, 48, 64, 48, 1);

  // ---- low_freq ----
  k_zero_u32<<<1, 32, 0, stream>>>(FLAG);
  k_maxflag<<<g1d(P1), 256, 0, stream>>>(img, FLAG, (int)P1);
  k_prescale<<<g1d(P1), 256, 0, stream>>>(img, XN, FLAG, (int)P1);
  // opp chain
  conv(XN,   opwh[0], op_b[0], OPP1, nullptr,   3, 256, 192, 2,  32, 0, BN_S);
  conv(OPP1, opwh[1], op_b[1], OPP2, nullptr,  32, 128,  96, 2,  64, 0, BN_S);
  conv(OPP2, opwh[2], op_b[2], OPP3, nullptr,  64,  64,  48, 2, 128, 0, BN_S);
  conv(OPP3, opwh[3], op_b[3], OPP4, nullptr, 128,  32,  24, 2, 128, 0, 1.f);
  conv(OPP4, opwh[4], op_b[4], OPP5, nullptr, 128,  16,  12, 2, 128, 0, 1.f);
  k_opp_head<<<64, 256, 0, stream>>>(OPP5, op_w[5], op_b[5], PR);
  // exposure + local gamma + bilateral
  k_expmul<<<g1d(P1), 256, 0, stream>>>(XN, PR, EXPO, B, 3 * H * W);
  conv(EXPO, lgwh[0], lg_b[0], LGA, nullptr, 3, H, W, 1, 3, 0, 1.f);
  conv(LGA,  lgwh[1], lg_b[1], LGB, nullptr, 3, H, W, 1, 3, 0, 1.f);
  conv(LGB,  lgwh[2], lg_b[2], LGC, nullptr, 3, H, W, 1, 3, 0, 1.f);
  conv(LGC,  lgwh[3], lg_b[3], LGD, LGC,     3, H, W, 1, 3, 0, 1.f);  // d = conv(c)+c
  conv(LGD,  lgwh[4], lg_b[4], LGC, LGB,     3, H, W, 1, 3, 0, 1.f);  // e = conv(d)+b
  conv(LGC,  lgwh[5], lg_b[5], LGD, LGA,     3, H, W, 1, 3, 2, 1.f);  // sigmoid(conv(e)+a)
  k_mul<<<g1d(P1), 256, 0, stream>>>(EXPO, LGD, GAMMA, P1);
  k_bilateral<<<g1d(P1), 256, 0, stream>>>(GAMMA, PR, BILO, B, 3, H, W);
  // ll = down^3(bilateral)
  k_blur5<<<g1d(294912L), 256, 0, stream>>>(BILO, LL1, nullptr, nullptr, 1.f, B, 3, 256, 192, 128, 96, 2);
  k_blur5<<<g1d(73728L), 256, 0, stream>>>(LL1, LL2, nullptr, nullptr, 1.f, B, 3, 128, 96, 64, 48, 2);
  k_blur5<<<g1d(18432L), 256, 0, stream>>>(LL2, LL3, nullptr, nullptr, 1.f, B, 3, 64, 48, 32, 24, 2);

  // ---- high_freq: three MLD blocks ----
  auto mld = [&](const float* in, f16* const* wh, const float* const* bb,
                 int Hl, int Wl, float* yout) {
    int N = Hl * Wl;
    conv(in,  wh[0], bb[0], A24, nullptr,  3, Hl, Wl, 1, 24, 1, 1.f);   // x = relu
    conv(A24, wh[1], bb[1], U3,  nullptr, 24, Hl, Wl, 1,  3, 1, 1.f);   // u = relu
    conv(A24, wh[2], bb[2], B24, nullptr, 24, Hl, Wl, 1, 24, 1, 1.f);   // f = relu
    k_einsum_v<<<8 * 72, 256, 0, stream>>>(U3, B24, VV, N);
    k_apply_v<<<g1d(8L * 24 * N), 256, 0, stream>>>(VV, U3, A24, N);     // res (reuse A24)
    conv(A24, wh[3], bb[3], yout, nullptr, 24, Hl, Wl, 1, 3, 0, 1.f);
  };
  const float* m1b[4] = {m1_b[0], m1_b[1], m1_b[2], m1_b[3]};
  const float* m2b[4] = {m2_b[0], m2_b[1], m2_b[2], m2_b[3]};
  mld(H1, m1wh, m1b, 256, 192, Y1);
  mld(H2, m2wh, m2b, 128,  96, Y2);
  mld(H3, m1wh, m1b,  64,  48, Y3);

  // ---- csf: concat [self, rz(other), rz(other)] (9ch) -> conv -> HH ----
  // level 1
  k_resize<<<g1d(P1), 256, 0, stream>>>(Y1, C1,                 B, 3, 256, 192, 256, 192, 9L * 256 * 192);
  k_resize<<<g1d(P1), 256, 0, stream>>>(Y2, C1 + 3L * 256 * 192, B, 3, 128,  96, 256, 192, 9L * 256 * 192);
  k_resize<<<g1d(P1), 256, 0, stream>>>(Y3, C1 + 6L * 256 * 192, B, 3,  64,  48, 256, 192, 9L * 256 * 192);
  conv(C1, csfwh, csf_b, HH1, nullptr, 9, 256, 192, 1, 3, 0, 1.f);
  // level 2
  k_resize<<<g1d(294912L), 256, 0, stream>>>(Y2, C2,               B, 3, 128, 96, 128, 96, 9L * 128 * 96);
  k_resize<<<g1d(294912L), 256, 0, stream>>>(Y1, C2 + 3L * 128 * 96, B, 3, 256, 192, 128, 96, 9L * 128 * 96);
  k_resize<<<g1d(294912L), 256, 0, stream>>>(Y3, C2 + 6L * 128 * 96, B, 3, 64, 48, 128, 96, 9L * 128 * 96);
  conv(C2, csfwh, csf_b, HH2, nullptr, 9, 128, 96, 1, 3, 0, 1.f);
  // level 3
  k_resize<<<g1d(73728L), 256, 0, stream>>>(Y3, C3,               B, 3, 64, 48, 64, 48, 9L * 64 * 48);
  k_resize<<<g1d(73728L), 256, 0, stream>>>(Y2, C3 + 3L * 64 * 48, B, 3, 128, 96, 64, 48, 9L * 64 * 48);
  k_resize<<<g1d(73728L), 256, 0, stream>>>(Y1, C3 + 6L * 64 * 48, B, 3, 256, 192, 64, 48, 9L * 64 * 48);
  conv(C3, csfwh, csf_b, HH3, nullptr, 9, 64, 48, 1, 3, 0, 1.f);

  // ---- reconstruct: rec = up(rec) + hh, then + img ----
  float* REC3 = G2;   // dead, exact size 8*3*64*48
  float* REC2 = G1;   // dead, exact size 8*3*128*96
  k_resize<<<g1d(73728L), 256, 0, stream>>>(LL3, TMPF, B, 3, 32, 24, 64, 48, 3L * 64 * 48);
  k_blur5<<<g1d(73728L), 256, 0, stream>>>(TMPF, REC3, HH3, nullptr, 1.f, B, 3, 64, 48, 64, 48, 1);
  k_resize<<<g1d(294912L), 256, 0, stream>>>(REC3, TMPF, B, 3, 64, 48, 128, 96, 3L * 128 * 96);
  k_blur5<<<g1d(294912L), 256, 0, stream>>>(TMPF, REC2, HH2, nullptr, 1.f, B, 3, 128, 96, 128, 96, 1);
  k_resize<<<g1d(P1), 256, 0, stream>>>(REC2, TMPF, B, 3, 128, 96, 256, 192, 3L * 256 * 192);
  // out = blur(up(rec2)) + hh1 + img
  k_blur5<<<g1d(P1), 256, 0, stream>>>(TMPF, (float*)d_out, HH1, img, 1.f, B, 3, 256, 192, 256, 192, 1);
}
